// SpikeDialogueModel_77721728188878
// MI455X (gfx1250) — matrix operand order repaired and checked by reading, not compiled, not run on hardware
//
#include <hip/hip_runtime.h>
#include <hip/hip_bf16.h>

// ---- vector types ----
typedef _Float16 v16h  __attribute__((ext_vector_type(16)));
typedef _Float16 half8 __attribute__((ext_vector_type(8)));
typedef float    v8f   __attribute__((ext_vector_type(8)));
typedef float    f4v   __attribute__((ext_vector_type(4)));

#define B_  8
#define S_  16
#define V_  32000
#define H_  512
#define E_  256
#define T_  20

// ---------------------------------------------------------------------------
// W2 (V,H) fp32 -> f16  (once per launch; 32.8 MB -> L2-resident thereafter)
// ---------------------------------------------------------------------------
__global__ void __launch_bounds__(256) w2_convert(const float* __restrict__ W2,
                                                  _Float16* __restrict__ W2h) {
  size_t i = ((size_t)blockIdx.x * 256 + threadIdx.x) * 4;
  f4v f = *(const f4v*)&W2[i];
  W2h[i + 0] = (_Float16)f[0];
  W2h[i + 1] = (_Float16)f[1];
  W2h[i + 2] = (_Float16)f[2];
  W2h[i + 3] = (_Float16)f[3];
}

// ---------------------------------------------------------------------------
// cur1[b][s][h] = sum_e embed[x[b,s]][e] * W1[h][e] + b1[h]
// one block per token (128 tokens), fp32 (feeds the threshold cascade)
// ---------------------------------------------------------------------------
__global__ void __launch_bounds__(256) cur1_kernel(const int* __restrict__ x,
                                                   const float* __restrict__ embed,
                                                   const float* __restrict__ W1,
                                                   const float* __restrict__ b1,
                                                   float* __restrict__ cur1) {
  __shared__ float se[E_];
  const int bs = blockIdx.x;              // bs = b*S + s
  const int token = x[bs];
  se[threadIdx.x] = embed[(size_t)token * E_ + threadIdx.x];
  __syncthreads();
  for (int h = threadIdx.x; h < H_; h += 256) {
    const float* w = &W1[(size_t)h * E_];
    float acc = b1[h];
    #pragma unroll 8
    for (int e = 0; e < E_; ++e) acc = fmaf(se[e], w[e], acc);
    cur1[(size_t)bs * H_ + h] = acc;
  }
}

// ---------------------------------------------------------------------------
// Fully fused S x T recurrence — ONE kernel launch for all 320 steps.
// No cross-block dependency exists: each block owns 128 output columns of V
// (its syn2/mem2 state lives in VGPRs for the whole run) and redundantly
// maintains mem1 (16 floats/thread in VGPRs) from the shared cur1 tensor.
// Per step: LDS spike matrix -> 16x v_wmma_f32_16x16x32_f16 per wave ->
// register-resident Synaptic update. W2h streamed from L2 every step.
// ---------------------------------------------------------------------------
__global__ void __launch_bounds__(256) snn_fused(
    const float* __restrict__ cur1,        // (B,S,H)
    const _Float16* __restrict__ W2h,      // (V,H) f16
    const float* __restrict__ b2,          // (V)
    float* __restrict__ outs,              // (B,S,V)
    float* __restrict__ out_mem1,          // (B,H)
    float* __restrict__ out_syn2,          // (B,V)
    float* __restrict__ out_mem2,          // (B,V)
    const float* __restrict__ p_beta1,
    const float* __restrict__ p_alpha2,
    const float* __restrict__ p_beta2) {
  __shared__ _Float16 sA[16 * H_];         // 16 KB padded A matrix (16 x 512)
  const int tid = threadIdx.x;

  const float b1c = fminf(fmaxf(p_beta1[0],  0.f), 1.f);
  const float a2c = fminf(fmaxf(p_alpha2[0], 0.f), 1.f);
  const float b2c = fminf(fmaxf(p_beta2[0],  0.f), 1.f);

  // rows 8..15 of A are permanent zero padding (written once)
  for (int idx = tid; idx < B_ * H_; idx += 256)
    sA[B_ * H_ + idx] = (_Float16)0.0f;

  const int lane    = tid & 31;
  const int wave    = tid >> 5;
  const int n       = lane & 15;
  const int halfSel = lane >> 4;                          // K-half select
  const int vcol    = blockIdx.x * 128 + wave * 16 + n;   // owned column in V
  const _Float16* __restrict__ Brow = &W2h[(size_t)vcol * H_];
  const float bias = b2[vcol];

  // register-resident recurrent state
  float m1[16];                                           // mem1 slice
  #pragma unroll
  for (int j = 0; j < 16; ++j) m1[j] = 0.0f;
  v8f syn = {}, mem = {};                                 // syn2/mem2 (lanes 0-15)

  for (int s = 0; s < S_; ++s) {
    for (int t = 0; t < T_; ++t) {
      // guard sA overwrite vs previous step's fragment reads
      __syncthreads();
      // Leaky update (replicated per block) + spike matrix into LDS
      #pragma unroll
      for (int j = 0; j < 16; ++j) {
        const int idx = tid + 256 * j;
        const int b = idx >> 9, h = idx & (H_ - 1);
        float m = m1[j];
        const float reset = (m > 0.8f) ? 0.8f : 0.0f;     // detached reset
        m = b1c * m + cur1[((size_t)b * S_ + s) * H_ + h] - reset;
        m1[j] = m;
        sA[b * H_ + h] = (m > 0.8f) ? (_Float16)1.0f : (_Float16)0.0f;
      }
      __syncthreads();

      // cur2 tile = spk1 @ W2h^T  (K=512 -> 16 WMMAs)
      v8f c = {};
      #pragma unroll 4
      for (int kk = 0; kk < 16; ++kk) {
        const int k0 = kk * 32;
        // A fragment (16-bit 16x32 layout): lanes 0-15: K 0..7,16..23; 16-31:+8
        half8 alo = *(const half8*)&sA[n * H_ + k0 + halfSel * 8];
        half8 ahi = *(const half8*)&sA[n * H_ + k0 + 16 + halfSel * 8];
        v16h a = __builtin_shufflevector(alo, ahi,
                  0,1,2,3,4,5,6,7,8,9,10,11,12,13,14,15);
        // B fragment: lanes 0-15 hold K 0..7,16..23 of column n; 16-31: +8
        const _Float16* bp = Brow + k0 + halfSel * 8;
        half8 blo = *(const half8*)bp;
        half8 bhi = *(const half8*)(bp + 16);
        v16h bfrag = __builtin_shufflevector(blo, bhi,
                  0,1,2,3,4,5,6,7,8,9,10,11,12,13,14,15);
        c = __builtin_amdgcn_wmma_f32_16x16x32_f16(
                false, a, false, bfrag, (short)0, c, false, false);
      }

      // Synaptic update on register state. f32 C/D layout: lane n (0-15),
      // VGPR r holds row M=r -> c[0..7] = cur2[b=0..7][vcol].
      if (lane < 16) {
        #pragma unroll
        for (int b = 0; b < B_; ++b) {
          const float cur2  = c[b] + bias;
          syn[b] = a2c * syn[b] + cur2;
          const float reset = (mem[b] > 1.0f) ? 1.0f : 0.0f; // detached reset
          mem[b] = b2c * mem[b] + syn[b] - reset;
        }
        if (t == T_ - 1) {
          #pragma unroll
          for (int b = 0; b < B_; ++b)
            outs[((size_t)b * S_ + s) * V_ + vcol] = (mem[b] > 1.0f) ? 1.0f : 0.0f;
        }
      }
    }
  }

  // final states
  if (lane < 16) {
    #pragma unroll
    for (int b = 0; b < B_; ++b) {
      out_syn2[(size_t)b * V_ + vcol] = syn[b];
      out_mem2[(size_t)b * V_ + vcol] = mem[b];
    }
  }
  if (blockIdx.x == 0) {
    #pragma unroll
    for (int j = 0; j < 16; ++j) out_mem1[tid + 256 * j] = m1[j];
  }
}

// ---------------------------------------------------------------------------
extern "C" void kernel_launch(void* const* d_in, const int* in_sizes, int n_in,
                              void* d_out, int out_size, void* d_ws, size_t ws_size,
                              hipStream_t stream) {
  const int*   x      = (const int*)  d_in[0];
  const float* embed  = (const float*)d_in[1];
  const float* W1     = (const float*)d_in[2];
  const float* b1     = (const float*)d_in[3];
  const float* W2     = (const float*)d_in[4];
  const float* b2     = (const float*)d_in[5];
  const float* beta1  = (const float*)d_in[6];
  const float* alpha2 = (const float*)d_in[7];
  const float* beta2  = (const float*)d_in[8];

  // d_out = [outs (B,S,V) | mem1 (B,H) | syn2 (B,V) | mem2 (B,V)]
  float* outs     = (float*)d_out;
  float* out_mem1 = outs + (size_t)B_ * S_ * V_;
  float* out_syn2 = out_mem1 + (size_t)B_ * H_;
  float* out_mem2 = out_syn2 + (size_t)B_ * V_;

  // workspace: W2h (f16, 32.8 MB) | cur1 (256 KB)
  char* ws = (char*)d_ws;
  _Float16* W2h  = (_Float16*)ws;  ws += (size_t)V_ * H_ * sizeof(_Float16);
  float*    cur1 = (float*)ws;

  w2_convert<<<(V_ * H_) / 1024, 256, 0, stream>>>(W2, W2h);
  cur1_kernel<<<B_ * S_, 256, 0, stream>>>(x, embed, W1, b1, cur1);
  snn_fused<<<V_ / 128, 256, 0, stream>>>(
      cur1, W2h, b2, outs, out_mem1, out_syn2, out_mem2,
      beta1, alpha2, beta2);
}
